// MultiInstRefiner_69896297775506
// MI455X (gfx1250) — compile-verified, matrix operand order checked by reading, not run on hardware
//
#include <hip/hip_runtime.h>
#include <hip/hip_bf16.h>
#include <math.h>

typedef __attribute__((ext_vector_type(16))) _Float16 v16h;
typedef __attribute__((ext_vector_type(8)))  _Float16 v8h;
typedef __attribute__((ext_vector_type(8)))  float    v8f;

#define KNUM 6
#define HH 512
#define WW 512
#define WIN 128
#define NPIX (WIN*WIN)

static __device__ __forceinline__ int imin(int a, int b) { return a < b ? a : b; }

// ---------------------------------------------------------------------------
// Box selection: distance = |1 - sum_k pred[k,0] - mean_k pred[k,2]| sampled
// every 10 px on a 51x51 grid, argmax (first occurrence), clamp 128-window.
// ---------------------------------------------------------------------------
__global__ void box_kernel(const float* __restrict__ pred, int* __restrict__ box) {
    __shared__ float sv[256];
    __shared__ int   si[256];
    int tid = threadIdx.x;
    float best = -1e30f;
    int bidx = 0x7fffffff;
    for (int i = tid; i < 51 * 51; i += 256) {
        int iy = i / 51, ix = i % 51;
        int y = iy * 10, x = ix * 10;
        float T = 0.f, B = 0.f;
        for (int k = 0; k < KNUM; ++k) {
            T += pred[(((size_t)k * 3 + 0) * HH + y) * WW + x];
            B += pred[(((size_t)k * 3 + 2) * HH + y) * WW + x];
        }
        B *= (1.f / (float)KNUM);
        float d = fabsf(1.f - T - B);
        if (d > best) { best = d; bidx = i; }   // ascending i -> first occurrence kept
    }
    sv[tid] = best; si[tid] = bidx;
    __syncthreads();
    for (int s = 128; s > 0; s >>= 1) {
        if (tid < s) {
            if (sv[tid + s] > sv[tid] ||
                (sv[tid + s] == sv[tid] && si[tid + s] < si[tid])) {
                sv[tid] = sv[tid + s]; si[tid] = si[tid + s];
            }
        }
        __syncthreads();
    }
    if (tid == 0) {
        int idx = si[0];
        int cx = (idx % 51) * 10, cy = (idx / 51) * 10;
        int x1 = cx - WIN / 2, y1 = cy - WIN / 2;
        int x2 = cx + WIN / 2, y2 = cy + WIN / 2;
        int ox = -imin(x1, 0) + imin(WW - x1, 0) - imin(x2, 0) + imin(WW - x2, 0);
        int oy = -imin(y1, 0) + imin(HH - y1, 0) - imin(y2, 0) + imin(HH - y2, 0);
        box[0] = y1 + oy;   // y1
        box[1] = x1 + ox;   // x1
    }
}

// ---------------------------------------------------------------------------
// Weight repack to WMMA A-fragment order: dst[mt][kk][cc][lane*16+h] holds
// W[oc = mt*16 + lane%16][ic = cc*32 + Klocal(lane,h)][kh][kw], zero-padded.
// Klocal = (h&7) + 16*(h>=8) + 8*(lane>=16)  (16-bit A/B operand layout).
// ---------------------------------------------------------------------------
__global__ void repack_kernel(const float* __restrict__ w, _Float16* __restrict__ dst,
                              int OC, int IC, int ICC, int MT) {
    int idx = blockIdx.x * blockDim.x + threadIdx.x;
    int total = MT * 9 * ICC * 512;
    if (idx >= total) return;
    int e = idx & 511;
    int rest = idx >> 9;
    int cc = rest % ICC; rest /= ICC;
    int kk = rest % 9;
    int mt = rest / 9;
    int lane = e >> 4, h = e & 15;
    int M = lane & 15;
    int kl = (h & 7) + ((h >> 3) << 4) + ((lane >> 4) << 3);
    int oc = mt * 16 + M;
    int ic = cc * 32 + kl;
    float v = 0.f;
    if (oc < OC && ic < IC) v = w[((size_t)oc * IC + ic) * 9 + kk];
    dst[idx] = (_Float16)v;
}

// ---------------------------------------------------------------------------
// Patch extraction (once): featT[k][y][x][64] f16 from feat_list, and x-image
// into fcatT channels 96..98 (99..127 zeroed).
// ---------------------------------------------------------------------------
__global__ void extract_static_kernel(const float* __restrict__ x,
                                      const float* __restrict__ feat,
                                      const int* __restrict__ box,
                                      _Float16* __restrict__ featT,
                                      _Float16* __restrict__ fcatT) {
    int idx = blockIdx.x * blockDim.x + threadIdx.x;
    if (idx >= KNUM * NPIX) return;
    int xx = idx % WIN;
    int yy = (idx / WIN) % WIN;
    int k  = idx / NPIX;
    int gy = box[0] + yy, gx = box[1] + xx;
    size_t pix = (size_t)idx;
    for (int c = 0; c < 64; ++c)
        featT[pix * 64 + c] =
            (_Float16)feat[(((size_t)k * 64 + c) * HH + gy) * WW + gx];
    for (int c = 0; c < 3; ++c)
        fcatT[pix * 128 + 96 + c] =
            (_Float16)x[(((size_t)k * 3 + c) * HH + gy) * WW + gx];
    for (int c = 99; c < 128; ++c)
        fcatT[pix * 128 + c] = (_Float16)0.f;
}

// Per-iteration: predT[k][y][x][4] f16 from current pl (d_out).
__global__ void extract_pred_kernel(const float* __restrict__ pl,
                                    const int* __restrict__ box,
                                    _Float16* __restrict__ predT) {
    int idx = blockIdx.x * blockDim.x + threadIdx.x;
    if (idx >= KNUM * NPIX) return;
    int xx = idx % WIN;
    int yy = (idx / WIN) % WIN;
    int k  = idx / NPIX;
    int gy = box[0] + yy, gx = box[1] + xx;
    size_t pix = (size_t)idx;
    for (int c = 0; c < 3; ++c)
        predT[pix * 4 + c] =
            (_Float16)pl[(((size_t)k * 3 + c) * HH + gy) * WW + gx];
    predT[pix * 4 + 3] = (_Float16)0.f;
}

// ---------------------------------------------------------------------------
// Implicit-GEMM 3x3 conv (pad=1 on the 128x128 patch) using WMMA f16->f32.
//   ICC: input-channel chunks of 32 (input stride = ICC*32, f16 ch-fastest)
//   MW:  waves along M (OC), NW: waves along N (pixels); MW*NW == 8
//   R:   output rows per block (stages R+2 rows; A-fragment reused R times,
//        R independent accumulator chains hide WMMA->WMMA RAW latency)
//   ACT: 0 = none (f16 out), 1 = leaky 0.2 (f16 out), 2 = f32 planar (head)
//   GROUPED: split conv; blockIdx.y = group g; input = featT * predT[...,g]
// grid = ((WIN/R) * (WIN/(NW*16)), groups, K)
// ---------------------------------------------------------------------------
template <int ICC, int MW, int NW, int R, int ACT, bool GROUPED>
__global__ __launch_bounds__(256) void conv3x3_wmma(
    const _Float16* __restrict__ inT,
    const _Float16* __restrict__ predT,
    const _Float16* __restrict__ wpack,   // [MTglobal][9][ICC][512]
    const float* __restrict__ bias,
    _Float16* __restrict__ outT,          // f16 ch-fastest, stride ocs
    float* __restrict__ outF32,           // planar f32 (ACT==2)
    int ocs) {
    constexpr int PIX  = NW * 16;
    constexpr int COLS = PIX + 2;
    constexpr int ROWS = R + 2;
    constexpr int ICS  = ICC * 32;

    __shared__ _Float16 lds[ROWS * COLS * 32];

    const int tid  = threadIdx.x;
    const int wid  = tid >> 5;       // wave32
    const int lane = tid & 31;
    const int m = wid / NW;
    const int n = wid % NW;

    const int tilesX = WIN / PIX;
    const int y0 = (blockIdx.x / tilesX) * R;
    const int x0 = (blockIdx.x % tilesX) * PIX;
    const int k  = blockIdx.z;
    const int mtb = blockIdx.y * MW;   // M-tile base (group offset for split)

    v8f acc[R];
#pragma unroll
    for (int r = 0; r < R; ++r)
#pragma unroll
        for (int e = 0; e < 8; ++e) acc[r][e] = 0.f;

    const _Float16* wbase = wpack + ((size_t)(mtb + m) * 9) * ICC * 512 + lane * 16;

    for (int cc = 0; cc < ICC; ++cc) {
        __syncthreads();
        // ---- stage ROWS rows x COLS cols x 32 ic into LDS (zero conv pad) ----
        const int NU = ROWS * COLS * 4;   // 16-byte units
        for (int u = tid; u < NU; u += 256) {
            int pos = u >> 2, icq = u & 3;
            int row = pos / COLS, col = pos % COLS;
            int yy = y0 + row - 1, xx = x0 + col - 1;
            v8h val;
#pragma unroll
            for (int e = 0; e < 8; ++e) val[e] = (_Float16)0.f;
            if (yy >= 0 && yy < WIN && xx >= 0 && xx < WIN) {
                size_t pix = ((size_t)k * WIN + yy) * WIN + xx;
                val = *(const v8h*)(inT + pix * ICS + cc * 32 + icq * 8);
                if (GROUPED) {
                    _Float16 s = predT[pix * 4 + blockIdx.y];
#pragma unroll
                    for (int e = 0; e < 8; ++e) val[e] = val[e] * s;
                }
            }
            *(v8h*)&lds[pos * 32 + icq * 8] = val;
        }
        __syncthreads();

        // ---- 9 taps: A from packed weights (reused across R rows), B from
        //      LDS, R independent WMMA accumulation chains ----
#pragma unroll
        for (int kk = 0; kk < 9; ++kk) {
            const int kh = kk / 3, kw = kk % 3;
            v16h a = *(const v16h*)(wbase + ((size_t)kk * ICC + cc) * 512);
            // prefetch next A fragment (global_prefetch path)
            if (!(kk == 8 && cc == ICC - 1)) {
                const int nkk = (kk == 8) ? 0 : kk + 1;
                const int ncc = (kk == 8) ? cc + 1 : cc;
                __builtin_prefetch(wbase + ((size_t)nkk * ICC + ncc) * 512, 0, 0);
            }

            const int col = n * 16 + (lane & 15) + kw;
#pragma unroll
            for (int r = 0; r < R; ++r) {
                const int base = ((kh + r) * COLS + col) * 32 + ((lane >> 4) << 3);
                v8h b0 = *(const v8h*)&lds[base];        // K = {0..7}   (+8 hi)
                v8h b1 = *(const v8h*)&lds[base + 16];   // K = {16..23} (+8 hi)
                v16h b;
#pragma unroll
                for (int e = 0; e < 8; ++e) { b[e] = b0[e]; b[8 + e] = b1[e]; }
                acc[r] = __builtin_amdgcn_wmma_f32_16x16x32_f16(
                    false, a, false, b, (short)0, acc[r], false, false);
            }
        }
    }

    // ---- epilogue: bias + activation + store ----
    const int px = x0 + n * 16 + (lane & 15);
#pragma unroll
    for (int r = 0; r < R; ++r) {
        const int y = y0 + r;
        const size_t pix = ((size_t)k * WIN + y) * WIN + px;
#pragma unroll
        for (int e = 0; e < 8; ++e) {
            int oc = (mtb + m) * 16 + e + ((lane >> 4) << 3);
            if (ACT == 2) {
                if (oc < 3) {
                    float v = acc[r][e] + bias[oc];
                    outF32[(((size_t)k * 3 + oc) * WIN + y) * WIN + px] = v;
                }
            } else {
                float v = acc[r][e] + bias[oc];
                if (ACT == 1) v = v > 0.f ? v : 0.2f * v;
                outT[pix * ocs + oc] = (_Float16)v;
            }
        }
    }
}

// ---------------------------------------------------------------------------
// Cross-instance combine: fsT[k][pix][96] -> fcatT[k][pix][0..95]
// feat_t = S1/5 - S0 + 2 f0 ; feat_r = S0 - f0 + f1 ; feat_b = S2/6
// ---------------------------------------------------------------------------
__global__ void combine_kernel(const _Float16* __restrict__ fsT,
                               _Float16* __restrict__ fcatT) {
    int idx = blockIdx.x * blockDim.x + threadIdx.x;
    if (idx >= NPIX * 32) return;
    int c = idx & 31;
    int pix = idx >> 5;
    float f0[KNUM], f1[KNUM];
    float S0 = 0.f, S1 = 0.f, S2 = 0.f;
#pragma unroll
    for (int k = 0; k < KNUM; ++k) {
        size_t base = ((size_t)k * NPIX + pix) * 96;
        float a = (float)fsT[base + c];
        float b = (float)fsT[base + 32 + c];
        float d = (float)fsT[base + 64 + c];
        f0[k] = a; f1[k] = b;
        S0 += a; S1 += b; S2 += d;
    }
    float fb = S2 * (1.f / (float)KNUM);
    float t1 = S1 * (1.f / (float)(KNUM - 1)) - S0;
#pragma unroll
    for (int k = 0; k < KNUM; ++k) {
        size_t base = ((size_t)k * NPIX + pix) * 128;
        fcatT[base + c]      = (_Float16)(t1 + 2.f * f0[k]);
        fcatT[base + 32 + c] = (_Float16)(S0 - f0[k] + f1[k]);
        fcatT[base + 64 + c] = (_Float16)fb;
    }
}

// (tanh(pr)+1)/2 on inner 108x108, scattered back into pl (d_out).
__global__ void update_kernel(const float* __restrict__ pr,
                              const int* __restrict__ box,
                              float* __restrict__ pl) {
    int idx = blockIdx.x * blockDim.x + threadIdx.x;
    if (idx >= KNUM * 3 * 108 * 108) return;
    int xx = idx % 108;
    int t = idx / 108;
    int yy = t % 108; t /= 108;
    int c = t % 3;
    int k = t / 3;
    float v = pr[(((size_t)k * 3 + c) * WIN + 10 + yy) * WIN + 10 + xx];
    v = (tanhf(v) + 1.f) * 0.5f;
    pl[(((size_t)k * 3 + c) * HH + box[0] + 10 + yy) * WW + box[1] + 10 + xx] = v;
}

// ---------------------------------------------------------------------------
extern "C" void kernel_launch(void* const* d_in, const int* in_sizes, int n_in,
                              void* d_out, int out_size, void* d_ws, size_t ws_size,
                              hipStream_t stream) {
    const float* x_in      = (const float*)d_in[0];
    const float* pred_list = (const float*)d_in[1];
    const float* feat_list = (const float*)d_in[2];
    const float* split_w   = (const float*)d_in[3];
    const float* split_b   = (const float*)d_in[4];
    const float* merge1_w  = (const float*)d_in[5];
    const float* merge1_b  = (const float*)d_in[6];
    const float* merge2_w  = (const float*)d_in[7];
    const float* merge2_b  = (const float*)d_in[8];
    const float* head_w    = (const float*)d_in[9];
    const float* head_b    = (const float*)d_in[10];
    float* pl = (float*)d_out;

    // workspace layout (256B aligned)
    char* ws = (char*)d_ws;
    size_t off = 0;
    auto take = [&](size_t bytes) { size_t o = off; off += (bytes + 255) & ~(size_t)255; return o; };
    int*       box    = (int*)      (ws + take(16));
    _Float16*  featT  = (_Float16*) (ws + take((size_t)KNUM * NPIX * 64 * 2));
    _Float16*  fcatT  = (_Float16*) (ws + take((size_t)KNUM * NPIX * 128 * 2));
    _Float16*  predT  = (_Float16*) (ws + take((size_t)KNUM * NPIX * 4 * 2));
    _Float16*  fsT    = (_Float16*) (ws + take((size_t)KNUM * NPIX * 96 * 2));
    _Float16*  h1T    = (_Float16*) (ws + take((size_t)KNUM * NPIX * 64 * 2));
    _Float16*  fmT    = (_Float16*) (ws + take((size_t)KNUM * NPIX * 64 * 2));
    float*     prBuf  = (float*)    (ws + take((size_t)KNUM * 3 * NPIX * 4));
    _Float16*  wpS    = (_Float16*) (ws + take((size_t)6 * 9 * 2 * 512 * 2));
    _Float16*  wpM1   = (_Float16*) (ws + take((size_t)4 * 9 * 4 * 512 * 2));
    _Float16*  wpM2   = (_Float16*) (ws + take((size_t)4 * 9 * 2 * 512 * 2));
    _Float16*  wpH    = (_Float16*) (ws + take((size_t)1 * 9 * 2 * 512 * 2));

    // pl <- pred_list
    hipMemcpyAsync(pl, pred_list, (size_t)KNUM * 3 * HH * WW * sizeof(float),
                   hipMemcpyDeviceToDevice, stream);

    box_kernel<<<1, 256, 0, stream>>>(pred_list, box);

    extract_static_kernel<<<(KNUM * NPIX + 255) / 256, 256, 0, stream>>>(
        x_in, feat_list, box, featT, fcatT);

    // weight repacks: (OC, IC, ICC, MT)
    repack_kernel<<<(6 * 9 * 2 * 512 + 255) / 256, 256, 0, stream>>>(split_w,  wpS,  96, 64, 2, 6);
    repack_kernel<<<(4 * 9 * 4 * 512 + 255) / 256, 256, 0, stream>>>(merge1_w, wpM1, 64, 99, 4, 4);
    repack_kernel<<<(4 * 9 * 2 * 512 + 255) / 256, 256, 0, stream>>>(merge2_w, wpM2, 64, 64, 2, 4);
    repack_kernel<<<(1 * 9 * 2 * 512 + 255) / 256, 256, 0, stream>>>(head_w,   wpH,   3, 64, 2, 1);

    for (int it = 0; it < 2; ++it) {
        extract_pred_kernel<<<(KNUM * NPIX + 255) / 256, 256, 0, stream>>>(pl, box, predT);

        // split: grouped, IC=64 (2 chunks), OC 32/group, PIX=64, R=2
        conv3x3_wmma<2, 2, 4, 2, 0, true><<<dim3((WIN / 2) * (WIN / 64), 3, KNUM), 256, 0, stream>>>(
            featT, predT, wpS, split_b, fsT, nullptr, 96);

        combine_kernel<<<(NPIX * 32 + 255) / 256, 256, 0, stream>>>(fsT, fcatT);

        // merge1: IC 99->128 (4 chunks), OC=64, PIX=32, R=2, leaky 0.2
        conv3x3_wmma<4, 4, 2, 2, 1, false><<<dim3((WIN / 2) * (WIN / 32), 1, KNUM), 256, 0, stream>>>(
            fcatT, nullptr, wpM1, merge1_b, h1T, nullptr, 64);

        // merge2: IC=64, OC=64, PIX=32, R=2
        conv3x3_wmma<2, 4, 2, 2, 0, false><<<dim3((WIN / 2) * (WIN / 32), 1, KNUM), 256, 0, stream>>>(
            h1T, nullptr, wpM2, merge2_b, fmT, nullptr, 64);

        // head: IC=64, OC 3 (padded 16), PIX=128, R=2, f32 planar out
        conv3x3_wmma<2, 1, 8, 2, 2, false><<<dim3((WIN / 2) * (WIN / 128), 1, KNUM), 256, 0, stream>>>(
            fmT, nullptr, wpH, head_b, nullptr, prBuf, 0);

        update_kernel<<<(KNUM * 3 * 108 * 108 + 255) / 256, 256, 0, stream>>>(prBuf, box, pl);
    }
}
